// MultiheadedAttention_19559281066467
// MI455X (gfx1250) — compile-verified, matrix operand order checked by reading, not run on hardware
//
#include <hip/hip_runtime.h>
#include <hip/hip_bf16.h>
#include <math.h>
#include <stdint.h>

// Problem constants (from reference): B=8, S=1024, D=1024, H=16, HD=64
#define BQ  8
#define SQ  1024
#define DQ  1024
#define HQ  16
#define HDQ 64
#define D3Q 3072
#define MROWS (BQ * SQ)   // 8192

#define XS_STRIDE 40      // padded LDS row stride for 32-element rows
#define KS_STRIDE 72      // padded LDS row stride for 64-element rows

typedef __attribute__((ext_vector_type(16))) __bf16 v16bf;
typedef __attribute__((ext_vector_type(8)))  float  v8f;
typedef __attribute__((ext_vector_type(4)))  unsigned int v4u;
typedef __attribute__((ext_vector_type(8)))  int    v8i;
typedef __attribute__((ext_vector_type(4)))  int    v4i;

#if defined(__has_builtin)
#  if __has_builtin(__builtin_amdgcn_tensor_load_to_lds)
#    define HAVE_TDM 1
#  endif
#  if __has_builtin(__builtin_amdgcn_s_wait_tensorcnt)
#    define HAVE_WTC 1
#  endif
#endif
#ifndef HAVE_TDM
#define HAVE_TDM 0
#endif
#ifndef HAVE_WTC
#define HAVE_WTC 0
#endif

union FragBF { v16bf v; unsigned int u[8]; };

// K-pair base for 16-bit WMMA operand VGPR v, lane-half h (CDNA5 A/B layout)
__device__ __forceinline__ int kpair(int v, int h) {
    return ((v & 4) ? 16 : 0) + (v & 3) * 2 + h * 8;
}

// Load a 16x32 bf16 WMMA operand fragment from a K-contiguous row-major
// matrix (global or LDS; generic pointer lowers to the right MEM ops).
__device__ __forceinline__ v16bf load_frag_row(const __bf16* src,
                                               int row0, int ld, int kbase, int lane) {
    FragBF f;
    int m = lane & 15, h = lane >> 4;
    const __bf16* p = src + (size_t)(row0 + m) * ld + kbase;
#pragma unroll
    for (int v = 0; v < 8; ++v)
        f.u[v] = *(const unsigned int*)(p + kpair(v, h));
    return f.v;
}

__device__ __forceinline__ v8f wmma_bf16(v16bf a, v16bf b, v8f c) {
    return __builtin_amdgcn_wmma_f32_16x16x32_bf16(
        false, a, false, b, (short)0, c, false, false);
}

__device__ __forceinline__ float red16_max(float x) {
#pragma unroll
    for (int m = 1; m <= 8; m <<= 1) x = fmaxf(x, __shfl_xor(x, m, 32));
    return x;
}
__device__ __forceinline__ float red16_sum(float x) {
#pragma unroll
    for (int m = 1; m <= 8; m <<= 1) x += __shfl_xor(x, m, 32);
    return x;
}

// ---- CDNA5 async copy: global -> LDS, 16B per lane (ASYNCcnt tracked) ----
__device__ __forceinline__ void async_b128(unsigned lds_off, const void* g) {
    asm volatile("global_load_async_to_lds_b128 %0, %1, off"
                 :: "v"(lds_off), "v"((unsigned long long)(uintptr_t)g)
                 : "memory");
}
__device__ __forceinline__ void wait_async0() {
    asm volatile("s_wait_asynccnt 0x0" ::: "memory");
}
__device__ __forceinline__ void wait_tensor0() {
#if HAVE_WTC
    __builtin_amdgcn_s_wait_tensorcnt(0);
#else
    asm volatile("s_wait_tensorcnt 0x0" ::: "memory");
#endif
}

// ---- Tensor Data Mover descriptor (D#) per CDNA5 ISA 08_async_tensor §8 ----
struct TdmDesc { v4u g0; v8i g1; v4i g2; v4i g3; };

__device__ __forceinline__ void tdm_build(TdmDesc& d, unsigned lds_off,
        unsigned long long gaddr, unsigned td0, unsigned td1, unsigned stride0,
        unsigned tile0, unsigned tile1, unsigned pad_int, unsigned pad_amt) {
    d.g0[0] = 1u;                                      // count=1 (valid D#)
    d.g0[1] = lds_off;                                 // lds_addr (bytes)
    d.g0[2] = (unsigned)gaddr;                         // global_addr[31:0]
    d.g0[3] = ((unsigned)(gaddr >> 32) & 0x01FFFFFFu)  // global_addr[56:32]
            | (2u << 30);                              // type = 2 ("image")
    d.g1[0] = (int)((1u << 16)                         // data_size = 2 bytes
            | (1u << 20)                               // pad_enable
            | (pad_int << 22) | (pad_amt << 25));
    d.g1[1] = (int)(td0 << 16);                        // tensor_dim0[15:0]
    d.g1[2] = (int)((td0 >> 16) | (td1 << 16));        // td0[31:16] | td1[15:0]
    d.g1[3] = (int)((td1 >> 16) | (tile0 << 16));      // td1[31:16] | tile_dim0
    d.g1[4] = (int)(tile1 & 0xFFFFu);                  // tile_dim1 (tile_dim2=0)
    d.g1[5] = (int)stride0;                            // tensor_dim0_stride[31:0]
    d.g1[6] = 0;                                       // stride0[47:32]|stride1 lo
    d.g1[7] = 0;
#pragma unroll
    for (int i = 0; i < 4; ++i) { d.g2[i] = 0; d.g3[i] = 0; }
}

__device__ __forceinline__ void tdm_issue(const TdmDesc& d) {
#if HAVE_TDM
#if __clang_major__ >= 23
    v8i z;
#pragma unroll
    for (int i = 0; i < 8; ++i) z[i] = 0;
    __builtin_amdgcn_tensor_load_to_lds(d.g0, d.g1, d.g2, d.g3, z, 0);
#else
    __builtin_amdgcn_tensor_load_to_lds(d.g0, d.g1, d.g2, d.g3, 0);
#endif
#else
    (void)d;
#endif
}

// ---------------------------------------------------------------- converts
__global__ void cvt_f32_bf16(const float* __restrict__ src, __bf16* __restrict__ dst, long n) {
    long i = (long)blockIdx.x * blockDim.x + threadIdx.x;
    long stride = (long)gridDim.x * blockDim.x;
    for (; i < n; i += stride) dst[i] = (__bf16)src[i];
}

// ---------------------------------------------------------------- GEMM core
// WG tile 128(M) x 64(N), K-step 32. X tile staged via async-to-LDS (all
// threads, 2x b128 each); W tile staged via TDM (wave 0) with pad so the
// DMA itself writes the bank-padded layout. Fragments then read from LDS.
__device__ __forceinline__ void gemm_tiles(
    const __bf16* __restrict__ A, const __bf16* __restrict__ W,
    unsigned wTotalRows, __bf16* xs, __bf16* wsb,
    int r0blk, int c0blk, int wm, int wn, v8f acc[2][2])
{
    int tid  = threadIdx.x;
    int lane = tid & 31;
    int wv   = tid >> 5;
    unsigned xs_off = (unsigned)(uintptr_t)xs;   // generic low 32 = LDS offset
    unsigned ws_off = (unsigned)(uintptr_t)wsb;

#pragma unroll 1
    for (int kk = 0; kk < DQ; kk += 32) {
        if (kk + 32 < DQ)   // L2 prefetch hint for next X K-tile
            __builtin_prefetch(A + (size_t)(r0blk + (tid & 127)) * DQ + kk + 32, 0, 1);
        // X tile: 128 rows x 32 elems = 512 16B-chunks, 2 per thread
#pragma unroll
        for (int i = 0; i < 2; ++i) {
            int c = tid + 256 * i;
            int row = c >> 2, ch = c & 3;
            async_b128(xs_off + (unsigned)(row * XS_STRIDE + ch * 8) * 2,
                       A + (size_t)(r0blk + row) * DQ + kk + ch * 8);
        }
#if HAVE_TDM
        if (wv == 0) {   // W tile 64 rows x 32 elems via Tensor Data Mover
            TdmDesc d;
            tdm_build(d, ws_off,
                      (unsigned long long)(uintptr_t)(W + (size_t)c0blk * DQ + kk),
                      DQ, wTotalRows, DQ, /*tile0=*/32, /*tile1=*/64,
                      /*pad_int=*/3 /*16 dw=64B rows*/, /*pad_amt=*/3 /*4 dw*/);
            tdm_issue(d);
            wait_tensor0();
        }
#else
        {   // fallback: W tile via async copies, 1 chunk per thread
            int row = tid >> 2, ch = tid & 3;
            async_b128(ws_off + (unsigned)(row * XS_STRIDE + ch * 8) * 2,
                       W + (size_t)(c0blk + row) * DQ + kk + ch * 8);
        }
#endif
        wait_async0();
        __syncthreads();

        v16bf a0 = load_frag_row(xs,  wm * 32,      XS_STRIDE, 0, lane);
        v16bf a1 = load_frag_row(xs,  wm * 32 + 16, XS_STRIDE, 0, lane);
        v16bf b0 = load_frag_row(wsb, wn * 32,      XS_STRIDE, 0, lane);
        v16bf b1 = load_frag_row(wsb, wn * 32 + 16, XS_STRIDE, 0, lane);
        acc[0][0] = wmma_bf16(a0, b0, acc[0][0]);
        acc[0][1] = wmma_bf16(a0, b1, acc[0][1]);
        acc[1][0] = wmma_bf16(a1, b0, acc[1][0]);
        acc[1][1] = wmma_bf16(a1, b1, acc[1][1]);
        __syncthreads();
    }
}

// ---------------------------------------------------------------- QKV GEMM
__global__ __launch_bounds__(256) void gemm_qkv(
    const __bf16* __restrict__ X, const __bf16* __restrict__ W,
    const float*  __restrict__ bias,
    __bf16* __restrict__ qb, __bf16* __restrict__ kb, __bf16* __restrict__ vtb)
{
    __shared__ __align__(16) __bf16 xs[128 * XS_STRIDE];
    __shared__ __align__(16) __bf16 wsb[64 * XS_STRIDE];
    int lane = threadIdx.x & 31;
    int wv   = threadIdx.x >> 5;
    int wm = wv & 3, wn = wv >> 2;
    int r0blk = blockIdx.y * 128, c0blk = blockIdx.x * 64;

    v8f acc[2][2] = {};
    gemm_tiles(X, W, D3Q, xs, wsb, r0blk, c0blk, wm, wn, acc);

    int r0 = r0blk + wm * 32, c0 = c0blk + wn * 32;
    int n = lane & 15, hl = lane >> 4;
#pragma unroll
    for (int i = 0; i < 2; ++i)
#pragma unroll
    for (int j = 0; j < 2; ++j) {
        int col  = c0 + j * 16 + n;
        float bi = bias[col];
        int head = col / (3 * HDQ);
        int cc   = col - head * (3 * HDQ);
        int which = cc >> 6;          // 0=q 1=k 2=v
        int hd    = cc & 63;
#pragma unroll
        for (int r = 0; r < 8; ++r) {
            int row  = r0 + i * 16 + r + hl * 8;
            int bidx = row >> 10;
            int sidx = row & 1023;
            __bf16 bv = (__bf16)(acc[i][j][r] + bi);
            if (which == 0)
                qb[((size_t)(bidx * HQ + head) * SQ + sidx) * HDQ + hd] = bv;
            else if (which == 1)
                kb[((size_t)(bidx * HQ + head) * SQ + sidx) * HDQ + hd] = bv;
            else
                vtb[((size_t)(bidx * HQ + head) * HDQ + hd) * SQ + sidx] = bv;
        }
    }
}

// ---------------------------------------------------------------- attention
// 8 waves/block all share one (b,h) and march over the same KV blocks, so K
// and V^T 32-row tiles are TDM-staged once per block per step. P re-swizzle
// (C-layout -> A-frag) goes through per-wave LDS guarded by s_wait_dscnt.
__global__ __launch_bounds__(256) void flash_attn(
    const __bf16* __restrict__ qb, const __bf16* __restrict__ kb,
    const __bf16* __restrict__ vtb, const float* __restrict__ mask,
    __bf16* __restrict__ ob)
{
    __shared__ __align__(16) __bf16 kts[32 * KS_STRIDE];   // K tile (s x hd)
    __shared__ __align__(16) __bf16 vts[64 * XS_STRIDE];   // V^T tile (hd x s)
    __shared__ __align__(16) __bf16 plds[8][16 * 32];

    int tid  = threadIdx.x;
    int lane = tid & 31;
    int wv   = tid >> 5;
    int gw   = blockIdx.x * 8 + wv;
    int qt   = gw & 63;
    int bh   = gw >> 6;
    int s0   = qt * 16;

    const __bf16* qp = qb  + (size_t)bh * SQ * HDQ;
    const __bf16* kp = kb  + (size_t)bh * SQ * HDQ;
    const __bf16* vp = vtb + (size_t)bh * HDQ * SQ;
    __bf16*       op = ob  + (size_t)bh * SQ * HDQ;

    unsigned kts_off = (unsigned)(uintptr_t)kts;
    unsigned vts_off = (unsigned)(uintptr_t)vts;

    v16bf qa0 = load_frag_row(qp, s0, HDQ, 0,  lane);
    v16bf qa1 = load_frag_row(qp, s0, HDQ, 32, lane);

    float mi[8], li[8];
#pragma unroll
    for (int r = 0; r < 8; ++r) { mi[r] = -3.0e38f; li[r] = 0.0f; }
    v8f acco[4] = {};

    int n = lane & 15, hl = lane >> 4;

#pragma unroll 1
    for (int j0 = 0; j0 < SQ; j0 += 32) {
#if HAVE_TDM
        if (wv == 0) {
            TdmDesc dk, dv;
            // K tile: 32 rows x 64 elems (row=128B -> pad_int=4, +16B pad)
            tdm_build(dk, kts_off, (unsigned long long)(uintptr_t)(kp + (size_t)j0 * HDQ),
                      HDQ, SQ, HDQ, /*tile0=*/64, /*tile1=*/32, 4, 3);
            // V^T tile: 64 rows x 32 elems (row=64B -> pad_int=3, +16B pad)
            tdm_build(dv, vts_off, (unsigned long long)(uintptr_t)(vp + j0),
                      SQ, HDQ, SQ, /*tile0=*/32, /*tile1=*/64, 3, 3);
            tdm_issue(dk);
            tdm_issue(dv);
            wait_tensor0();
        }
#else
        {   // K: 32 rows x 8 chunks; V^T: 64 rows x 4 chunks (1 each / thread)
            int row = tid >> 3, ch = tid & 7;
            async_b128(kts_off + (unsigned)(row * KS_STRIDE + ch * 8) * 2,
                       kp + (size_t)(j0 + row) * HDQ + ch * 8);
            row = tid >> 2; ch = tid & 3;
            async_b128(vts_off + (unsigned)(row * XS_STRIDE + ch * 8) * 2,
                       vp + (size_t)row * SQ + j0 + ch * 8);
            wait_async0();
        }
#endif
        __syncthreads();

        float pt[2][8];
#pragma unroll
        for (int t = 0; t < 2; ++t) {
            v16bf b0 = load_frag_row(kts, t * 16, KS_STRIDE, 0,  lane);
            v16bf b1 = load_frag_row(kts, t * 16, KS_STRIDE, 32, lane);
            v8f sc = {};
            sc = wmma_bf16(qa0, b0, sc);
            sc = wmma_bf16(qa1, b1, sc);
            int col = j0 + t * 16 + n;
#pragma unroll
            for (int r = 0; r < 8; ++r) {
                int row = s0 + r + hl * 8;
                pt[t][r] = sc[r] * 0.125f + mask[(size_t)row * SQ + col];
            }
        }
        float fac[8];
#pragma unroll
        for (int r = 0; r < 8; ++r) {
            float mcur = red16_max(fmaxf(pt[0][r], pt[1][r]));
            float mnew = fmaxf(mi[r], mcur);
            fac[r]   = __expf(mi[r] - mnew);
            float p0 = __expf(pt[0][r] - mnew);
            float p1 = __expf(pt[1][r] - mnew);
            pt[0][r] = p0; pt[1][r] = p1;
            li[r] = li[r] * fac[r] + red16_sum(p0 + p1);
            mi[r] = mnew;
        }
#pragma unroll
        for (int c = 0; c < 4; ++c)
#pragma unroll
            for (int r = 0; r < 8; ++r) acco[c][r] *= fac[r];

        // P -> LDS -> bf16 A-fragment (same-wave re-swizzle)
#pragma unroll
        for (int t = 0; t < 2; ++t)
#pragma unroll
        for (int r = 0; r < 8; ++r)
            plds[wv][(r + hl * 8) * 32 + t * 16 + n] = (__bf16)pt[t][r];
        asm volatile("s_wait_dscnt 0" ::: "memory");
        FragBF pf;
#pragma unroll
        for (int v = 0; v < 8; ++v)
            pf.u[v] = *(const unsigned int*)(&plds[wv][(lane & 15) * 32 + kpair(v, hl)]);
        asm volatile("" ::: "memory");

#pragma unroll
        for (int c = 0; c < 4; ++c) {
            v16bf vbf = load_frag_row(vts, c * 16, XS_STRIDE, 0, lane);
            acco[c] = wmma_bf16(pf.v, vbf, acco[c]);
        }
        __syncthreads();
    }

#pragma unroll
    for (int c = 0; c < 4; ++c)
#pragma unroll
    for (int r = 0; r < 8; ++r) {
        int row = s0 + r + hl * 8;
        op[(size_t)row * HDQ + c * 16 + n] = (__bf16)(acco[c][r] / li[r]);
    }
}

// ---------------------------------------------------------------- out proj
__global__ __launch_bounds__(256) void gemm_out(
    const __bf16* __restrict__ A, const __bf16* __restrict__ W,
    const float* __restrict__ bias, float* __restrict__ out)
{
    __shared__ __align__(16) __bf16 xs[128 * XS_STRIDE];
    __shared__ __align__(16) __bf16 wsb[64 * XS_STRIDE];
    int lane = threadIdx.x & 31;
    int wv   = threadIdx.x >> 5;
    int wm = wv & 3, wn = wv >> 2;
    int r0blk = blockIdx.y * 128, c0blk = blockIdx.x * 64;

    v8f acc[2][2] = {};
    gemm_tiles(A, W, DQ, xs, wsb, r0blk, c0blk, wm, wn, acc);

    int r0 = r0blk + wm * 32, c0 = c0blk + wn * 32;
    int n = lane & 15, hl = lane >> 4;
#pragma unroll
    for (int i = 0; i < 2; ++i)
#pragma unroll
    for (int j = 0; j < 2; ++j) {
        int col  = c0 + j * 16 + n;
        float bi = bias[col];
#pragma unroll
        for (int r = 0; r < 8; ++r) {
            int row = r0 + i * 16 + r + hl * 8;
            out[(size_t)row * DQ + col] = acc[i][j][r] + bi;
        }
    }
}

// ---------------------------------------------------------------- launch
extern "C" void kernel_launch(void* const* d_in, const int* in_sizes, int n_in,
                              void* d_out, int out_size, void* d_ws, size_t ws_size,
                              hipStream_t stream) {
    const float* x    = (const float*)d_in[0];
    const float* mask = (const float*)d_in[1];
    const float* Wqkv = (const float*)d_in[2];
    const float* bqkv = (const float*)d_in[3];
    const float* Wo   = (const float*)d_in[4];
    const float* bo   = (const float*)d_in[5];
    float* out = (float*)d_out;

    char*  ws  = (char*)d_ws;
    size_t off = 0;
    auto carve = [&](size_t bytes) -> char* {
        char* p = ws + off;
        off += (bytes + 255) & ~(size_t)255;
        return p;
    };
    __bf16* xb   = (__bf16*)carve((size_t)MROWS * DQ * 2);
    __bf16* wqb  = (__bf16*)carve((size_t)D3Q * DQ * 2);
    __bf16* wob  = (__bf16*)carve((size_t)DQ * DQ * 2);
    __bf16* qb   = (__bf16*)carve((size_t)BQ * HQ * SQ * HDQ * 2);
    __bf16* kb   = (__bf16*)carve((size_t)BQ * HQ * SQ * HDQ * 2);
    __bf16* vtb  = (__bf16*)carve((size_t)BQ * HQ * HDQ * SQ * 2);
    __bf16* valb = (__bf16*)carve((size_t)MROWS * DQ * 2);

    cvt_f32_bf16<<<2048, 256, 0, stream>>>(x,    xb,  (long)MROWS * DQ);
    cvt_f32_bf16<<<1024, 256, 0, stream>>>(Wqkv, wqb, (long)D3Q * DQ);
    cvt_f32_bf16<<<512,  256, 0, stream>>>(Wo,   wob, (long)DQ * DQ);

    dim3 g1(D3Q / 64, MROWS / 128);
    gemm_qkv<<<g1, 256, 0, stream>>>(xb, wqb, bqkv, qb, kb, vtb);

    flash_attn<<<dim3((BQ * HQ * (SQ / 16)) / 8), 256, 0, stream>>>(
        qb, kb, vtb, mask, valb);

    dim3 g2(DQ / 64, MROWS / 128);
    gemm_out<<<g2, 256, 0, stream>>>(valb, wob, bo, out);
}
